// graphic_cls_aggregation_29403346108963
// MI455X (gfx1250) — matrix operand order repaired and checked by reading, not compiled, not run on hardware
//
#include <hip/hip_runtime.h>
#include <hip/hip_bf16.h>
#include <math.h>

// ---------------------------------------------------------------------------
// Problem constants (from the reference)
// ---------------------------------------------------------------------------
#define BB   4
#define NC   256
#define NF   768
#define DD   768
#define NT   (NC + NF)          // 1024 tokens after concat
#define TOPK 128
#define LN_EPS 1e-5f

typedef __bf16  bf16_t;
typedef __attribute__((ext_vector_type(16))) __bf16 v16bf;
typedef __attribute__((ext_vector_type(8)))  float  v8f;

struct U128 { unsigned int u[4]; };   // plain 16-byte chunk for b128 loads

union Frag32B {
    v16bf v;
    U128  q[2];
};

// ---------------------------------------------------------------------------
// Kernel 1: transpose + convert a weight matrix W[Din][Dout] (f32, row major)
// into Wt[Dout][Din] (bf16, row major) so GEMM B-fragments are contiguous in K
// ---------------------------------------------------------------------------
__global__ __launch_bounds__(256)
void wtranspose_bf16(const float* __restrict__ W, bf16_t* __restrict__ Wt,
                     int Din, int Dout) {
    int i = blockIdx.x * blockDim.x + threadIdx.x;
    int total = Din * Dout;
    if (i >= total) return;
    int n = i % Dout;           // output dim
    int k = i / Dout;           // input dim
    Wt[(size_t)n * Din + k] = (bf16_t)W[i];
}

// ---------------------------------------------------------------------------
// Kernel 2: build x = concat(cls, feats) per batch, converted to bf16
// xb layout: [B][NT][D]
// ---------------------------------------------------------------------------
__global__ __launch_bounds__(256)
void concat_bf16(const float* __restrict__ cls, const float* __restrict__ feats,
                 bf16_t* __restrict__ xb) {
    size_t i = (size_t)blockIdx.x * blockDim.x + threadIdx.x;
    size_t total = (size_t)BB * NT * DD;
    if (i >= total) return;
    int d = (int)(i % DD);
    size_t r = i / DD;
    int b = (int)(r / NT);
    int t = (int)(r % NT);
    float v = (t < NC) ? cls[((size_t)b * NC + t) * DD + d]
                       : feats[((size_t)b * NF + (t - NC)) * DD + d];
    xb[i] = (bf16_t)v;
}

// ---------------------------------------------------------------------------
// Kernel 3: generic WMMA GEMM   C[M x N] = A[M x K] * Bt[N x K]^T (+ bias)
//   A, Bt bf16 row-major; C f32; optional leaky-relu; optional scaled bf16 copy
//   Each wave computes a 32x32 tile (2 A frags x 2 B frags -> 4 WMMAs/K-step,
//   2 b128 loads per WMMA). 8 waves/block cover a 32x256 output tile.
//   Fragment layouts per CDNA5 ISA 7.12.2 (wave32).
// ---------------------------------------------------------------------------
__global__ __launch_bounds__(256)
void gemm_bf16_wmma(const bf16_t* __restrict__ A, const bf16_t* __restrict__ Bt,
                    const float* __restrict__ bias,
                    float* __restrict__ C, bf16_t* __restrict__ Cb, float cbScale,
                    int N, int K, int lda, int ldb, int ldc,
                    long long sA, long long sB, long long sC, int act) {
    const int wave = threadIdx.x >> 5;
    const int lane = threadIdx.x & 31;
    const int half = lane >> 4;        // 0: lanes 0-15, 1: lanes 16-31
    const int lm   = lane & 15;

    const int bz   = blockIdx.z;
    const bf16_t* Ab = A  + (size_t)bz * sA;
    const bf16_t* Bb = Bt + (size_t)bz * sB;

    const int row0 = blockIdx.y * 32;
    const int n0   = blockIdx.x * 256 + wave * 32;
    if (n0 >= N) return;               // wave-uniform; never true with exact grids
    const int col0 = n0 + lm;

    // A fragments: lane holds row (row0+lm) / (row0+16+lm);
    //   element e -> K = (e<8?0:16)+8*half+(e&7)
    const bf16_t* ap0 = Ab + (size_t)(row0 + lm) * lda + 8 * half;
    const bf16_t* ap1 = ap0 + (size_t)16 * lda;
    // B fragments: lane holds column col0 / col0+16; element e -> K = (e<8?0:16)+8*half+(e&7)
    const bf16_t* bp0 = Bb + (size_t)col0 * ldb + 8 * half;
    const bf16_t* bp1 = bp0 + (size_t)16 * ldb;

    v8f acc00 = {}, acc01 = {}, acc10 = {}, acc11 = {};
    for (int k0 = 0; k0 < K; k0 += 32) {
        // Unconditional speculative prefetch of the next K-block into the
        // near (WGP) cache level; past-the-end prefetches are dropped by HW.
        __builtin_prefetch(ap0 + k0 + 32, 0, 3);
        __builtin_prefetch(ap1 + k0 + 32, 0, 3);
        __builtin_prefetch(bp0 + k0 + 32, 0, 3);
        __builtin_prefetch(bp1 + k0 + 32, 0, 3);
        Frag32B fa0, fa1, fb0, fb1;
        fa0.q[0] = *(const U128*)(ap0 + k0);        // K: k0+8h .. +7
        fa0.q[1] = *(const U128*)(ap0 + k0 + 16);   // K: k0+16+8h .. +7
        fa1.q[0] = *(const U128*)(ap1 + k0);
        fa1.q[1] = *(const U128*)(ap1 + k0 + 16);
        fb0.q[0] = *(const U128*)(bp0 + k0);        // K: k0+8h .. +7
        fb0.q[1] = *(const U128*)(bp0 + k0 + 16);   // K: k0+16+8h .. +7
        fb1.q[0] = *(const U128*)(bp1 + k0);
        fb1.q[1] = *(const U128*)(bp1 + k0 + 16);
        acc00 = __builtin_amdgcn_wmma_f32_16x16x32_bf16(
                    false, fa0.v, false, fb0.v, (short)0, acc00, false, false);
        acc01 = __builtin_amdgcn_wmma_f32_16x16x32_bf16(
                    false, fa0.v, false, fb1.v, (short)0, acc01, false, false);
        acc10 = __builtin_amdgcn_wmma_f32_16x16x32_bf16(
                    false, fa1.v, false, fb0.v, (short)0, acc10, false, false);
        acc11 = __builtin_amdgcn_wmma_f32_16x16x32_bf16(
                    false, fa1.v, false, fb1.v, (short)0, acc11, false, false);
    }

    const float bv0 = bias ? bias[col0]      : 0.0f;
    const float bv1 = bias ? bias[col0 + 16] : 0.0f;
    float*  Cp  = C + (size_t)bz * sC;
    bf16_t* Cbp = Cb ? (Cb + (size_t)bz * sC) : (bf16_t*)0;

#pragma unroll
    for (int r = 0; r < 8; ++r) {
        // D layout: VGPR r -> M = r + 8*half, N = lane&15
        int rowA = row0 + r + 8 * half;
        int rowB = rowA + 16;
        float v00 = acc00[r] + bv0;
        float v01 = acc01[r] + bv1;
        float v10 = acc10[r] + bv0;
        float v11 = acc11[r] + bv1;
        if (act == 1) {
            v00 = (v00 > 0.0f) ? v00 : 0.01f * v00;
            v01 = (v01 > 0.0f) ? v01 : 0.01f * v01;
            v10 = (v10 > 0.0f) ? v10 : 0.01f * v10;
            v11 = (v11 > 0.0f) ? v11 : 0.01f * v11;
        }
        Cp[(size_t)rowA * ldc + col0]      = v00;
        Cp[(size_t)rowA * ldc + col0 + 16] = v01;
        Cp[(size_t)rowB * ldc + col0]      = v10;
        Cp[(size_t)rowB * ldc + col0 + 16] = v11;
        if (Cbp) {
            Cbp[(size_t)rowA * ldc + col0]      = (bf16_t)(v00 * cbScale);
            Cbp[(size_t)rowA * ldc + col0 + 16] = (bf16_t)(v01 * cbScale);
            Cbp[(size_t)rowB * ldc + col0]      = (bf16_t)(v10 * cbScale);
            Cbp[(size_t)rowB * ldc + col0 + 16] = (bf16_t)(v11 * cbScale);
        }
    }
}

// ---------------------------------------------------------------------------
// Kernel 4: top-128 of 1024 per row via bitonic sort in LDS.
// Order within the top-k is irrelevant downstream (all reductions over k).
// ---------------------------------------------------------------------------
__global__ __launch_bounds__(256)
void topk128_kernel(const float* __restrict__ attn,
                    float* __restrict__ topw, int* __restrict__ topi) {
    __shared__ unsigned long long keys[NT];
    const int row = blockIdx.x;                    // 0 .. B*NC-1
    const int tid = threadIdx.x;
    const float* ap = attn + (size_t)row * NT;

    for (int i = tid; i < NT; i += 256) {
        unsigned u = __float_as_uint(ap[i]);
        u = (u & 0x80000000u) ? ~u : (u | 0x80000000u);   // sortable encoding
        keys[i] = ((unsigned long long)u << 32) | (unsigned)i;
    }
    __syncthreads();

    for (int size = 2; size <= NT; size <<= 1) {
        for (int stride = size >> 1; stride > 0; stride >>= 1) {
            for (int t = tid; t < NT / 2; t += 256) {
                int i = 2 * t - (t & (stride - 1));
                int j = i + stride;
                bool asc = ((i & size) == 0);
                unsigned long long a = keys[i], b = keys[j];
                bool sw = asc ? (a > b) : (a < b);
                if (sw) { keys[i] = b; keys[j] = a; }
            }
            __syncthreads();
        }
    }

    if (tid < TOPK) {
        unsigned long long kk = keys[NT - 1 - tid];       // largest 128
        unsigned u  = (unsigned)(kk >> 32);
        unsigned fb = (u & 0x80000000u) ? (u & 0x7fffffffu) : ~u;
        topw[(size_t)row * TOPK + tid] = __uint_as_float(fb);
        topi[(size_t)row * TOPK + tid] = (int)(kk & 0xffffffffu);
    }
}

// ---------------------------------------------------------------------------
// Shared-memory softmax over 128 values (256 threads participating in syncs)
// ---------------------------------------------------------------------------
__device__ __forceinline__ void softmax128(float* s, float* red, int tid) {
    if (tid < 128) red[tid] = s[tid];
    __syncthreads();
    for (int off = 64; off > 0; off >>= 1) {
        if (tid < off) red[tid] = fmaxf(red[tid], red[tid + off]);
        __syncthreads();
    }
    float m = red[0];
    __syncthreads();
    if (tid < 128) { float e = __expf(s[tid] - m); s[tid] = e; red[tid] = e; }
    __syncthreads();
    for (int off = 64; off > 0; off >>= 1) {
        if (tid < off) red[tid] += red[tid + off];
        __syncthreads();
    }
    float inv = 1.0f / red[0];
    __syncthreads();
    if (tid < 128) s[tid] *= inv;
    __syncthreads();
}

// ---------------------------------------------------------------------------
// Kernel 5: gated neighbor aggregation -> e_Nh [B*NC][D]
// One block (256 threads) per (b,n) row; each thread owns 3 elements of D.
// ---------------------------------------------------------------------------
__global__ __launch_bounds__(256)
void aggregate_kernel(const float* __restrict__ e_h, const float* __restrict__ e_t,
                      const float* __restrict__ topw, const int* __restrict__ topi,
                      float* __restrict__ e_Nh) {
    __shared__ float sp[TOPK];          // attention softmax p
    __shared__ float ska[TOPK];         // ka weights -> ka softmax
    __shared__ float red[128];
    __shared__ float wred[TOPK * 8];    // per-wave partial dots

    const int row  = blockIdx.x;        // 0 .. B*NC-1
    const int b    = row >> 8;          // NC = 256
    const int tid  = threadIdx.x;
    const int lane = tid & 31;
    const int wv   = tid >> 5;

    const float* ehp  = e_h + (size_t)row * DD;
    const float* etb  = e_t + (size_t)b * NT * DD;
    const int*   idxp = topi + (size_t)row * TOPK;

    if (tid < TOPK) sp[tid] = topw[(size_t)row * TOPK + tid];
    __syncthreads();
    softmax128(sp, red, tid);           // p = softmax(topk_weight)

    float ehl[3];
#pragma unroll
    for (int j = 0; j < 3; ++j) ehl[j] = ehp[tid + 256 * j];

    // ka_weight[k] = sum_d Nb * tanh(e_h + p*Nb + (1-p)*e_h)
    for (int k = 0; k < TOPK; ++k) {
        const float* nb = etb + (size_t)idxp[k] * DD;
        float pk   = sp[k];
        float part = 0.0f;
#pragma unroll
        for (int j = 0; j < 3; ++j) {
            float nv  = nb[tid + 256 * j];
            float ehv = ehl[j];
            float ehr = pk * nv + (1.0f - pk) * ehv;
            part += nv * tanhf(ehv + ehr);
        }
        for (int off = 16; off > 0; off >>= 1) part += __shfl_down(part, off, 32);
        if (lane == 0) wred[k * 8 + wv] = part;
    }
    __syncthreads();
    if (tid < TOPK) {
        float s = 0.0f;
        for (int j = 0; j < 8; ++j) s += wred[tid * 8 + j];
        ska[tid] = s;
    }
    __syncthreads();
    softmax128(ska, red, tid);          // ka_prob

    float acc[3] = {0.0f, 0.0f, 0.0f};
    for (int k = 0; k < TOPK; ++k) {
        const float* nb = etb + (size_t)idxp[k] * DD;
        float pk = ska[k];
#pragma unroll
        for (int j = 0; j < 3; ++j) acc[j] += pk * nb[tid + 256 * j];
    }
    float* outp = e_Nh + (size_t)row * DD;
#pragma unroll
    for (int j = 0; j < 3; ++j) outp[tid + 256 * j] = acc[j];
}

// ---------------------------------------------------------------------------
// Kernel 6: build the two bi-interaction GEMM inputs (bf16)
// in1 = (e_h + e_Nh)*0.1 + cls ; in2 = e_h*e_Nh*0.1 + cls
// ---------------------------------------------------------------------------
__global__ __launch_bounds__(256)
void build_bi_inputs(const float* __restrict__ cls, const float* __restrict__ e_h,
                     const float* __restrict__ e_Nh,
                     bf16_t* __restrict__ in1, bf16_t* __restrict__ in2) {
    size_t i = (size_t)blockIdx.x * blockDim.x + threadIdx.x;
    size_t total = (size_t)BB * NC * DD;
    if (i >= total) return;
    float c = cls[i], h = e_h[i], n = e_Nh[i];
    in1[i] = (bf16_t)((h + n) * 0.1f + c);
    in2[i] = (bf16_t)(h * n * 0.1f + c);
}

// ---------------------------------------------------------------------------
// Kernel 7: embedding = sum_emb + bi_emb, then LayerNorm per row -> d_out
// ---------------------------------------------------------------------------
__global__ __launch_bounds__(256)
void add_layernorm(const float* __restrict__ s1, const float* __restrict__ s2,
                   const float* __restrict__ lw, const float* __restrict__ lb,
                   float* __restrict__ out) {
    __shared__ float red[256];
    const int row = blockIdx.x;
    const int tid = threadIdx.x;
    const size_t base = (size_t)row * DD;

    float x[3];
    float sum = 0.0f, sq = 0.0f;
#pragma unroll
    for (int j = 0; j < 3; ++j) {
        int d = tid + 256 * j;
        x[j] = s1[base + d] + s2[base + d];
        sum += x[j];
        sq  += x[j] * x[j];
    }
    red[tid] = sum;
    __syncthreads();
    for (int off = 128; off > 0; off >>= 1) {
        if (tid < off) red[tid] += red[tid + off];
        __syncthreads();
    }
    float mu = red[0] * (1.0f / DD);
    __syncthreads();
    red[tid] = sq;
    __syncthreads();
    for (int off = 128; off > 0; off >>= 1) {
        if (tid < off) red[tid] += red[tid + off];
        __syncthreads();
    }
    float var = red[0] * (1.0f / DD) - mu * mu;
    float inv = rsqrtf(var + LN_EPS);
#pragma unroll
    for (int j = 0; j < 3; ++j) {
        int d = tid + 256 * j;
        out[base + d] = (x[j] - mu) * inv * lw[d] + lb[d];
    }
}

// ---------------------------------------------------------------------------
// Launch orchestration
// ---------------------------------------------------------------------------
extern "C" void kernel_launch(void* const* d_in, const int* in_sizes, int n_in,
                              void* d_out, int out_size, void* d_ws, size_t ws_size,
                              hipStream_t stream) {
    (void)in_sizes; (void)n_in; (void)out_size; (void)ws_size;

    const float* cls    = (const float*)d_in[0];
    const float* feats  = (const float*)d_in[1];
    const float* Whead  = (const float*)d_in[2];
    const float* bhead  = (const float*)d_in[3];
    const float* Wtail  = (const float*)d_in[4];
    const float* btail  = (const float*)d_in[5];
    const float* Wl1    = (const float*)d_in[6];
    const float* bl1    = (const float*)d_in[7];
    const float* Wl2    = (const float*)d_in[8];
    const float* bl2    = (const float*)d_in[9];
    const float* lnw    = (const float*)d_in[10];
    const float* lnb    = (const float*)d_in[11];
    float* out = (float*)d_out;

    // --- workspace carve-up (all 256B aligned) ---
    char* ws = (char*)d_ws;
    size_t off = 0;
    auto carve = [&](size_t bytes) -> void* {
        void* p = ws + off;
        off += (bytes + 255) & ~(size_t)255;
        return p;
    };
    bf16_t* wtH  = (bf16_t*)carve((size_t)DD * DD * 2);
    bf16_t* wtT  = (bf16_t*)carve((size_t)DD * DD * 2);
    bf16_t* wt1  = (bf16_t*)carve((size_t)DD * DD * 2);
    bf16_t* wt2  = (bf16_t*)carve((size_t)DD * DD * 2);
    bf16_t* xb   = (bf16_t*)carve((size_t)BB * NT * DD * 2);
    float*  e_h  = (float*) carve((size_t)BB * NC * DD * 4);
    bf16_t* e_hs = (bf16_t*)carve((size_t)BB * NC * DD * 2);   // scaled bf16 copy
    float*  e_t  = (float*) carve((size_t)BB * NT * DD * 4);
    bf16_t* e_tb = (bf16_t*)carve((size_t)BB * NT * DD * 2);
    float*  attn = (float*) carve((size_t)BB * NC * NT * 4);
    float*  topw = (float*) carve((size_t)BB * NC * TOPK * 4);
    int*    topi = (int*)   carve((size_t)BB * NC * TOPK * 4);
    float*  eNh  = (float*) carve((size_t)BB * NC * DD * 4);
    bf16_t* in1  = (bf16_t*)carve((size_t)BB * NC * DD * 2);
    bf16_t* in2  = (bf16_t*)carve((size_t)BB * NC * DD * 2);
    float*  semb = (float*) carve((size_t)BB * NC * DD * 4);
    float*  bemb = (float*) carve((size_t)BB * NC * DD * 4);

    const float attnScale = 1.0f / sqrtf((float)DD);

    // 1) weights -> bf16 transposed
    {
        int total = DD * DD;
        int g = (total + 255) / 256;
        wtranspose_bf16<<<g, 256, 0, stream>>>(Whead, wtH, DD, DD);
        wtranspose_bf16<<<g, 256, 0, stream>>>(Wtail, wtT, DD, DD);
        wtranspose_bf16<<<g, 256, 0, stream>>>(Wl1,   wt1, DD, DD);
        wtranspose_bf16<<<g, 256, 0, stream>>>(Wl2,   wt2, DD, DD);
    }
    // 2) x = concat(cls, feats) in bf16
    {
        size_t total = (size_t)BB * NT * DD;
        int g = (int)((total + 255) / 256);
        concat_bf16<<<g, 256, 0, stream>>>(cls, feats, xb);
    }
    // 3) e_h = cls @ Whead + b   (A = first 256 rows of each batch of xb)
    //    also writes scaled bf16 copy for the attention GEMM's A.
    gemm_bf16_wmma<<<dim3(DD / 256, NC / 32, BB), 256, 0, stream>>>(
        xb, wtH, bhead, e_h, e_hs, attnScale,
        DD, DD, DD, DD, DD,
        (long long)NT * DD, 0LL, (long long)NC * DD, 0);
    // 4) e_t = x @ Wtail + b  (flattened M = B*NT), plus bf16 copy for attn B
    gemm_bf16_wmma<<<dim3(DD / 256, (BB * NT) / 32, 1), 256, 0, stream>>>(
        xb, wtT, btail, e_t, e_tb, 1.0f,
        DD, DD, DD, DD, DD,
        0LL, 0LL, 0LL, 0);
    // 5) attn_logit[b] = (e_h*scale) @ e_t[b]^T   (Bt = e_tb rows are K-contiguous)
    gemm_bf16_wmma<<<dim3(NT / 256, NC / 32, BB), 256, 0, stream>>>(
        e_hs, e_tb, (const float*)0, attn, (bf16_t*)0, 1.0f,
        NT, DD, DD, DD, NT,
        (long long)NC * DD, (long long)NT * DD, (long long)NC * NT, 0);
    // 6) top-128 per row
    topk128_kernel<<<BB * NC, 256, 0, stream>>>(attn, topw, topi);
    // 7) gated aggregation -> e_Nh
    aggregate_kernel<<<BB * NC, 256, 0, stream>>>(e_h, e_t, topw, topi, eNh);
    // 8) bi-interaction inputs
    {
        size_t total = (size_t)BB * NC * DD;
        int g = (int)((total + 255) / 256);
        build_bi_inputs<<<g, 256, 0, stream>>>(cls, e_h, eNh, in1, in2);
    }
    // 9) sum_emb = leaky(lin1(in1)); bi_emb = leaky(lin2(in2))
    gemm_bf16_wmma<<<dim3(DD / 256, (BB * NC) / 32, 1), 256, 0, stream>>>(
        in1, wt1, bl1, semb, (bf16_t*)0, 1.0f,
        DD, DD, DD, DD, DD, 0LL, 0LL, 0LL, 1);
    gemm_bf16_wmma<<<dim3(DD / 256, (BB * NC) / 32, 1), 256, 0, stream>>>(
        in2, wt2, bl2, bemb, (bf16_t*)0, 1.0f,
        DD, DD, DD, DD, DD, 0LL, 0LL, 0LL, 1);
    // 10) add + LayerNorm -> output
    add_layernorm<<<BB * NC, 256, 0, stream>>>(semb, bemb, lnw, lnb, out);
}
